// GPTJAttention_19713899889368
// MI455X (gfx1250) — matrix operand order repaired and checked by reading, not compiled, not run on hardware
//
#include <hip/hip_runtime.h>

// ---------------------------------------------------------------------------
// GPT-J attention layer for MI455X (gfx1250, wave32, WMMA + TDM).
// Pipeline: f32->bf16 convert (+weight transpose) -> QKV GEMM (WMMA bf16,
// TDM double-buffered LDS staging, scatter epilogue to head-major q/k/v) ->
// RoPE -> flash attention (WMMA) -> output-projection GEMM (WMMA, f32 out).
// All matrix math: V_WMMA_F32_16X16X32_BF16 (16K FLOP/instr, f32 accum).
// GEMM tiles are DMA'd by the Tensor Data Mover (tensor_load_to_lds) with
// TENSORcnt-based double buffering when the builtin is available.
// ---------------------------------------------------------------------------

#define S_LEN  2048
#define HID    4096
#define NHEAD  16
#define DHEAD  256

typedef __attribute__((ext_vector_type(16))) __bf16 v16bf;
typedef __attribute__((ext_vector_type(8)))  float  v8f;
typedef __attribute__((ext_vector_type(4)))  unsigned int v4u;
typedef __attribute__((ext_vector_type(8)))  int v8i;
typedef __attribute__((ext_vector_type(4)))  int v4i;

#if defined(__has_builtin)
#if __has_builtin(__builtin_amdgcn_tensor_load_to_lds) && \
    __has_builtin(__builtin_amdgcn_s_wait_tensorcnt)
#define USE_TDM 1
#endif
#endif
#ifndef USE_TDM
#define USE_TDM 0
#endif

struct __align__(16) U4 { unsigned x, y, z, w; };
struct __align__(16) F4 { float x, y, z, w; };
struct __align__(8)  H4 { unsigned short a, b, c, d; };

union FragBF { v16bf v; U4 q[2]; unsigned short h[16]; };
union FAcc   { v8f  v; float f[8]; };

__device__ __forceinline__ unsigned short f2bf(float f) {
  union { float f; unsigned u; } x; x.f = f;
  unsigned r = x.u + 0x7FFFu + ((x.u >> 16) & 1u);   // round-to-nearest-even
  return (unsigned short)(r >> 16);
}
__device__ __forceinline__ float bf2f(unsigned short h) {
  union { unsigned u; float f; } x; x.u = ((unsigned)h) << 16; return x.f;
}

__device__ __forceinline__ v8f wmma_bf16(const v16bf& a, const v16bf& b, const v8f& c) {
  // (neg_a, A, neg_b, B, c_mod, C, reuse_a, reuse_b)
  return __builtin_amdgcn_wmma_f32_16x16x32_bf16(false, a, false, b, (short)0, c, false, false);
}

#if USE_TDM
// ---------------------------------------------------------------------------
// TDM: DMA a 128-row x 32-col bf16 tile (row pitch K elems in memory) into
// LDS with a 40-element (80 B) padded row pitch, via the Tensor Data Mover.
// D# bitfields per CDNA5 ISA ch.8: group0 = {ctl, lds_addr, global_addr lo,
// global_addr hi | type=2}; group1 packs data_size=2B, pad_enable,
// pad_interval=16 DWORDs, pad_amount=4 DWORDs, tensor/tile dims and stride.
// ---------------------------------------------------------------------------
__device__ __forceinline__ void tdm_load_tile(const unsigned short* gaddr,
                                              unsigned short* lds, int K) {
  unsigned long long ga = (unsigned long long)(size_t)gaddr;
  unsigned lds_off = (unsigned)(size_t)lds;            // LDS aperture: low 32b
  v4u g0;
  g0.x = 1u;                                           // count=1, user desc
  g0.y = lds_off;                                      // lds_addr (bytes)
  g0.z = (unsigned)ga;                                 // global_addr[31:0]
  g0.w = ((unsigned)(ga >> 32) & 0x01FFFFFFu) | (2u << 30);  // [56:32] | type=2
  v8i g1;
  g1[0] = (1 << 16)        // data_size = 2 bytes
        | (1 << 20)        // pad_enable (LDS dest padding)
        | (3 << 22)        // pad_interval: 16 DWORDs (one 64B tile row)
        | (3 << 25);       // pad_amount: 4 DWORDs (16 B) -> 80 B pitch
  g1[1] = (32 << 16);      // tensor_dim0 = 32 (bits 79:48, low half)
  g1[2] = (128 << 16);     // tensor_dim0 hi = 0, tensor_dim1 = 128
  g1[3] = (32 << 16);      // tensor_dim1 hi = 0, tile_dim0 = 32
  g1[4] = 128;             // tile_dim1 = 128, tile_dim2 = 0
  g1[5] = K;               // tensor_dim0_stride[31:0] (elements)
  g1[6] = 0;               // stride0 hi, tensor_dim1_stride lo
  g1[7] = 0;               // tensor_dim1_stride hi
  v4i g2 = {0, 0, 0, 0};   // 2D tile: groups 2/3 unused
  v4i g3 = {0, 0, 0, 0};
#if __clang_major__ >= 23
  v8i g4 = {0, 0, 0, 0, 0, 0, 0, 0};
  __builtin_amdgcn_tensor_load_to_lds(g0, g1, g2, g3, g4, 0);
#else
  __builtin_amdgcn_tensor_load_to_lds(g0, g1, g2, g3, 0);
#endif
}
#endif

// ---------------------------------------------------------------------------
// 1) f32 -> bf16 convert, 4 elements / thread
// ---------------------------------------------------------------------------
__global__ __launch_bounds__(256) void cvt4_kernel(const float* __restrict__ src,
                                                   unsigned short* __restrict__ dst, int n4) {
  int i = blockIdx.x * 256 + threadIdx.x;
  if (i >= n4) return;
  F4 f = *(const F4*)(src + (size_t)i * 4);
  H4 o; o.a = f2bf(f.x); o.b = f2bf(f.y); o.c = f2bf(f.z); o.d = f2bf(f.w);
  *(H4*)(dst + (size_t)i * 4) = o;
}

// ---------------------------------------------------------------------------
// 2) transpose + convert: src f32 [K][N] -> dst bf16 [N][K], 64x64 LDS tiles
// ---------------------------------------------------------------------------
__global__ __launch_bounds__(256) void tconv_kernel(const float* __restrict__ src,
                                                    unsigned short* __restrict__ dst,
                                                    int K, int N) {
  __shared__ unsigned short tile[64][65];
  int kb = blockIdx.y * 64, nb = blockIdx.x * 64;
  int t  = threadIdx.x;
  int tr = t >> 4;          // 0..15
  int tc = t & 15;          // 0..15 (x4 columns)
#pragma unroll
  for (int p = 0; p < 4; p++) {
    int r = tr + p * 16;    // k-row in tile
    F4 f = *(const F4*)(src + (size_t)(kb + r) * N + nb + tc * 4);
    tile[tc * 4 + 0][r] = f2bf(f.x);
    tile[tc * 4 + 1][r] = f2bf(f.y);
    tile[tc * 4 + 2][r] = f2bf(f.z);
    tile[tc * 4 + 3][r] = f2bf(f.w);
  }
  __syncthreads();
#pragma unroll
  for (int p = 0; p < 4; p++) {
    int nr = tr + p * 16;   // n-row in tile
    H4 o;
    o.a = tile[nr][tc * 4 + 0]; o.b = tile[nr][tc * 4 + 1];
    o.c = tile[nr][tc * 4 + 2]; o.d = tile[nr][tc * 4 + 3];
    *(H4*)(dst + (size_t)(nb + nr) * K + kb + tc * 4) = o;
  }
}

// ---------------------------------------------------------------------------
// 3) WMMA GEMM: C[M][N] = A[M][K] * BT[N][K]^T  (bf16 in, f32 acc)
//    Block tile 128x128x32, 8 waves arranged 4(M) x 2(N), each wave 32x64.
//    Double-buffered LDS staging: TDM (tensor_load_to_lds) when available,
//    overlapped with WMMA compute via s_wait_tensorcnt; manual b128 loads
//    otherwise. LDS rows padded to 40 bf16 (80 B pitch): conflict-free b128
//    fragment loads.
//    EPI=0: f32 row-major output.  EPI=1: bf16 scatter into q/k/v head-major.
// ---------------------------------------------------------------------------
template <int EPI>
__global__ __launch_bounds__(256) void gemm_bf16_kernel(
    const unsigned short* __restrict__ A,   // [M][K] bf16
    const unsigned short* __restrict__ BT,  // [N][K] bf16
    int M, int N, int K,
    float* __restrict__ Cout,
    unsigned short* __restrict__ qO, unsigned short* __restrict__ kO,
    unsigned short* __restrict__ vO) {
  __shared__ __align__(16) unsigned short As[2][128 * 40];
  __shared__ __align__(16) unsigned short Bs[2][128 * 40];

  const int tid  = threadIdx.x;
  const int lane = tid & 31, wid = tid >> 5;
  const int half = lane >> 4, ln = lane & 15;
  const int wm = wid >> 1, wn = wid & 1;
  const int mBase = blockIdx.y * 128, nBase = blockIdx.x * 128;

  FAcc acc[2][4];
#pragma unroll
  for (int mi = 0; mi < 2; mi++)
#pragma unroll
    for (int ni = 0; ni < 4; ni++)
#pragma unroll
      for (int r = 0; r < 8; r++) acc[mi][ni].f[r] = 0.0f;

  auto stageAB = [&](int buf, int kt) {
    const unsigned short* Ap = A  + (size_t)mBase * K + (size_t)kt * 32;
    const unsigned short* Bp = BT + (size_t)nBase * K + (size_t)kt * 32;
#if USE_TDM
    if (wid == 0) {                       // one wave drives the DMA engine
      tdm_load_tile(Ap, &As[buf][0], K);
      tdm_load_tile(Bp, &Bs[buf][0], K);
    }
#else
#pragma unroll
    for (int c = tid; c < 512; c += 256) {  // 128 rows x 4 b128-chunks
      int row = c >> 2, kc = c & 3;
      *(U4*)&As[buf][row * 40 + kc * 8] = *(const U4*)(Ap + (size_t)row * K + kc * 8);
      *(U4*)&Bs[buf][row * 40 + kc * 8] = *(const U4*)(Bp + (size_t)row * K + kc * 8);
    }
    __builtin_prefetch(Ap + 32, 0, 1);      // next k-tile -> global_prefetch_b8
    __builtin_prefetch(Bp + 32, 0, 1);
#endif
  };

  const int kSteps = K >> 5;
  stageAB(0, 0);                            // prologue: tile 0 in flight
  for (int kt = 0; kt < kSteps; kt++) {
    const int cur = kt & 1;
    const bool hasNext = (kt + 1) < kSteps;
    if (hasNext) stageAB(cur ^ 1, kt + 1);  // issue next tile's DMA first
#if USE_TDM
    if (wid == 0) {
      // in-order TENSORcnt: <=2 retires current tile, leaves next in flight
      if (hasNext) __builtin_amdgcn_s_wait_tensorcnt(2);
      else         __builtin_amdgcn_s_wait_tensorcnt(0);
    }
#endif
    __syncthreads();                        // current buffer valid for all waves

    FragBF af[2], bfrag[4];
#pragma unroll
    for (int mi = 0; mi < 2; mi++) {
      int row = wm * 32 + mi * 16 + ln;     // A-frag: row = lane%16
      af[mi].q[0] = *(const U4*)&As[cur][row * 40 + half * 8];       // k = half*8+0..7
      af[mi].q[1] = *(const U4*)&As[cur][row * 40 + 16 + half * 8];  // k = 16+half*8+0..7
    }
#pragma unroll
    for (int ni = 0; ni < 4; ni++) {
      int col = wn * 64 + ni * 16 + ln;     // B-frag: n = lane%16
      bfrag[ni].q[0] = *(const U4*)&Bs[cur][col * 40 + half * 8];    // k = half*8+0..7
      bfrag[ni].q[1] = *(const U4*)&Bs[cur][col * 40 + 16 + half * 8];
    }
#pragma unroll
    for (int mi = 0; mi < 2; mi++)
#pragma unroll
      for (int ni = 0; ni < 4; ni++)
        acc[mi][ni].v = wmma_bf16(af[mi].v, bfrag[ni].v, acc[mi][ni].v);
    __syncthreads();                        // done reading before buf reuse
  }

  // ---- epilogue: C row = half*8+r, col = lane%16 ----
  if (EPI == 1) {
#pragma unroll
    for (int mi = 0; mi < 2; mi++)
#pragma unroll
      for (int ni = 0; ni < 4; ni++) {
        int n = nBase + wn * 64 + ni * 16 + ln;  // 0..12287
        int which = n >> 12;                     // 0=q 1=k 2=v
        int hd = (n >> 8) & 15, dd = n & 255;
        unsigned short* dst = (which == 0) ? qO : ((which == 1) ? kO : vO);
#pragma unroll
        for (int r = 0; r < 8; r++) {
          int mrow = mBase + wm * 32 + mi * 16 + half * 8 + r;
          int bb = mrow >> 11, ss = mrow & 2047;
          dst[((size_t)(bb * NHEAD + hd) * S_LEN + ss) * DHEAD + dd] = f2bf(acc[mi][ni].f[r]);
        }
      }
  } else {
#pragma unroll
    for (int mi = 0; mi < 2; mi++)
#pragma unroll
      for (int ni = 0; ni < 4; ni++) {
        int n = nBase + wn * 64 + ni * 16 + ln;
#pragma unroll
        for (int r = 0; r < 8; r++) {
          int mrow = mBase + wm * 32 + mi * 16 + half * 8 + r;
          Cout[(size_t)mrow * N + n] = acc[mi][ni].f[r];
        }
      }
  }
}

// ---------------------------------------------------------------------------
// 4) GPT-J interleaved RoPE on first 64 dims of q and k (in place, bf16)
// ---------------------------------------------------------------------------
__global__ __launch_bounds__(256) void rope_kernel(const int* __restrict__ pid,
                                                   unsigned short* __restrict__ qB,
                                                   unsigned short* __restrict__ kB) {
  int idx = blockIdx.x * 256 + threadIdx.x;     // B*NH*S*32 threads
  int i  = idx & 31;
  int s  = (idx >> 5) & (S_LEN - 1);
  int bh = idx >> 16;
  int b  = bh >> 4;
  int pos = pid[b * S_LEN + s];
  // inv_freq = theta^(-2i/64);  ln(10000) = 9.210340372
  float freq = __expf(-(float)(2 * i) * (9.2103403720f / 64.0f));
  float ang  = (float)pos * freq;
  float sn, cs;
  __sincosf(ang, &sn, &cs);
  size_t base = ((size_t)bh * S_LEN + s) * DHEAD + 2 * i;
  {
    float x1 = bf2f(qB[base]), x2 = bf2f(qB[base + 1]);
    qB[base]     = f2bf(x1 * cs - x2 * sn);
    qB[base + 1] = f2bf(x2 * cs + x1 * sn);
  }
  {
    float x1 = bf2f(kB[base]), x2 = bf2f(kB[base + 1]);
    kB[base]     = f2bf(x1 * cs - x2 * sn);
    kB[base + 1] = f2bf(x2 * cs + x1 * sn);
  }
}

// ---------------------------------------------------------------------------
// 5) Flash attention: grid (S/64, B*NH), 4 waves/block, 1 wave = 16 q-rows.
//    Q: 8 A-frags held in VGPRs. K: B-frags from global (contiguous d).
//    V: 32x256 tile transposed into LDS (pitch 40) -> contiguous PV B-frags.
//    P: re-laid out C-frag -> A-frag through per-wave LDS scratch.
// ---------------------------------------------------------------------------
__global__ __launch_bounds__(128) void attn_kernel(const unsigned short* __restrict__ qB,
                                                   const unsigned short* __restrict__ kB,
                                                   const unsigned short* __restrict__ vB,
                                                   unsigned short* __restrict__ ctxB) {
  __shared__ __align__(16) unsigned short VT[256 * 40];  // [d][key], pitch 40
  __shared__ __align__(16) unsigned short Pl[4][512];    // per-wave 16x32 P tile

  const int tid  = threadIdx.x;
  const int lane = tid & 31, wid = tid >> 5;
  const int half = lane >> 4, ln = lane & 15;
  const int bh = blockIdx.y, b = bh >> 4, h = bh & 15;
  const int qb = blockIdx.x * 64 + wid * 16;

  const unsigned short* qp  = qB + ((size_t)bh * S_LEN + qb) * DHEAD;
  const unsigned short* kp0 = kB + (size_t)bh * S_LEN * DHEAD;
  const unsigned short* vp0 = vB + (size_t)bh * S_LEN * DHEAD;

  FragBF Qf[8];
#pragma unroll
  for (int c = 0; c < 8; c++) {
    const unsigned short* rp = qp + (size_t)ln * DHEAD + c * 32;
    Qf[c].q[0] = *(const U4*)(rp + half * 8);
    Qf[c].q[1] = *(const U4*)(rp + 16 + half * 8);
  }

  FAcc O[16];
#pragma unroll
  for (int dt = 0; dt < 16; dt++)
#pragma unroll
    for (int r = 0; r < 8; r++) O[dt].f[r] = 0.0f;
  float m[8], sum[8];
#pragma unroll
  for (int r = 0; r < 8; r++) { m[r] = -1e30f; sum[r] = 0.0f; }

  const int jEnd = blockIdx.x * 2 + 2;          // key blocks of 32 (causal bound)
  for (int j = 0; j < jEnd; j++) {
    const int kbase = j * 32;
    __syncthreads();
    // stage V tile transposed: VT[d][key]
    for (int c2 = tid; c2 < 1024; c2 += 128) {
      int key = c2 >> 5, dc = c2 & 31;
      U4 vv = *(const U4*)(vp0 + (size_t)(kbase + key) * DHEAD + dc * 8);
      const unsigned short* hh = (const unsigned short*)&vv;
#pragma unroll
      for (int i = 0; i < 8; i++) VT[(dc * 8 + i) * 40 + key] = hh[i];
    }
    __syncthreads();

    // ---- scores: S = Q (16x256) * K^T (256x32), two 16-key N-halves ----
    FAcc Sc0, Sc1;
#pragma unroll
    for (int r = 0; r < 8; r++) { Sc0.f[r] = 0.0f; Sc1.f[r] = 0.0f; }
    const unsigned short* kr0 = kp0 + (size_t)(kbase + ln) * DHEAD;
    const unsigned short* kr1 = kp0 + (size_t)(kbase + 16 + ln) * DHEAD;
#pragma unroll
    for (int c = 0; c < 8; c++) {
      FragBF Kf0, Kf1;
      Kf0.q[0] = *(const U4*)(kr0 + c * 32 + half * 8);
      Kf0.q[1] = *(const U4*)(kr0 + c * 32 + 16 + half * 8);
      Kf1.q[0] = *(const U4*)(kr1 + c * 32 + half * 8);
      Kf1.q[1] = *(const U4*)(kr1 + c * 32 + 16 + half * 8);
      Sc0.v = wmma_bf16(Qf[c].v, Kf0.v, Sc0.v);
      Sc1.v = wmma_bf16(Qf[c].v, Kf1.v, Sc1.v);
    }

    // ---- online softmax (rows live across 16 lanes of each half) ----
    float alpha_[8];
#pragma unroll
    for (int r = 0; r < 8; r++) {
      int rowg = qb + half * 8 + r;
      float v0 = Sc0.f[r] * 0.0625f;            // scale = HEAD_SIZE^-0.5
      float v1 = Sc1.f[r] * 0.0625f;
      if (kbase + ln > rowg)      v0 = -1e30f;  // causal mask
      if (kbase + 16 + ln > rowg) v1 = -1e30f;
      float vm = fmaxf(v0, v1);
      vm = fmaxf(vm, __shfl_xor(vm, 1, 32));
      vm = fmaxf(vm, __shfl_xor(vm, 2, 32));
      vm = fmaxf(vm, __shfl_xor(vm, 4, 32));
      vm = fmaxf(vm, __shfl_xor(vm, 8, 32));
      float mn = fmaxf(m[r], vm);
      float al = __expf(m[r] - mn);
      float p0 = __expf(v0 - mn);
      float p1 = __expf(v1 - mn);
      float rs = p0 + p1;
      rs += __shfl_xor(rs, 1, 32);
      rs += __shfl_xor(rs, 2, 32);
      rs += __shfl_xor(rs, 4, 32);
      rs += __shfl_xor(rs, 8, 32);
      sum[r] = sum[r] * al + rs;
      m[r] = mn;
      alpha_[r] = al;
      Pl[wid][(half * 8 + r) * 32 + ln]      = f2bf(p0);
      Pl[wid][(half * 8 + r) * 32 + 16 + ln] = f2bf(p1);
    }
#pragma unroll
    for (int dt = 0; dt < 16; dt++)
#pragma unroll
      for (int r = 0; r < 8; r++) O[dt].f[r] *= alpha_[r];

    // P C-layout -> A-frag via LDS (wave-private region, in-wave ordering)
    FragBF Pf;
    Pf.q[0] = *(const U4*)&Pl[wid][ln * 32 + half * 8];
    Pf.q[1] = *(const U4*)&Pl[wid][ln * 32 + 16 + half * 8];

    // ---- O += P (16x32) * V (32x256) ----
#pragma unroll
    for (int dt = 0; dt < 16; dt++) {
      FragBF Vf;
      Vf.q[0] = *(const U4*)&VT[(dt * 16 + ln) * 40 + half * 8];
      Vf.q[1] = *(const U4*)&VT[(dt * 16 + ln) * 40 + 16 + half * 8];
      O[dt].v = wmma_bf16(Pf.v, Vf.v, O[dt].v);
    }
  }

  // ---- normalize + write ctx bf16 [B,S,H] ----
#pragma unroll
  for (int r = 0; r < 8; r++) sum[r] = 1.0f / sum[r];
#pragma unroll
  for (int dt = 0; dt < 16; dt++)
#pragma unroll
    for (int r = 0; r < 8; r++) {
      int rowl = half * 8 + r;
      ctxB[(size_t)(b * S_LEN + qb + rowl) * HID + h * DHEAD + dt * 16 + ln] =
          f2bf(O[dt].f[r] * sum[r]);
    }
}

// ---------------------------------------------------------------------------
// launch:  ws layout (bytes, 288 MiB total):
//   hsB    @ 0          32M   bf16 [B*S][H]
//   wqkvT  @ 32M        96M   bf16 [3H][H]   (transposed)
//   woT    @ 128M       32M   bf16 [H][H]    (transposed)
//   qB     @ 160M       32M   bf16 [B,NH,S,D]
//   kB     @ 192M       32M
//   vB     @ 224M       32M
//   ctxB   @ 256M       32M   bf16 [B*S][H]
// ---------------------------------------------------------------------------
extern "C" void kernel_launch(void* const* d_in, const int* in_sizes, int n_in,
                              void* d_out, int out_size, void* d_ws, size_t ws_size,
                              hipStream_t stream) {
  (void)in_sizes; (void)n_in; (void)out_size; (void)ws_size;
  const int*   pid  = (const int*)d_in[0];
  const float* hs   = (const float*)d_in[1];
  const float* wqkv = (const float*)d_in[2];
  const float* wo   = (const float*)d_in[3];
  char* ws = (char*)d_ws;

  const size_t MB = 1024ull * 1024ull;
  unsigned short* hsB   = (unsigned short*)(ws + 0);
  unsigned short* wqkvT = (unsigned short*)(ws + 32 * MB);
  unsigned short* woT   = (unsigned short*)(ws + 128 * MB);
  unsigned short* qB    = (unsigned short*)(ws + 160 * MB);
  unsigned short* kB    = (unsigned short*)(ws + 192 * MB);
  unsigned short* vB    = (unsigned short*)(ws + 224 * MB);
  unsigned short* ctxB  = (unsigned short*)(ws + 256 * MB);
  float* out = (float*)d_out;

  // hidden_states: 2*2048*4096 = 16M f32 -> 4M vec4 threads
  cvt4_kernel<<<16384, 256, 0, stream>>>(hs, hsB, 4194304);
  // Wqkv [4096][12288] -> [12288][4096] bf16
  tconv_kernel<<<dim3(192, 64), 256, 0, stream>>>(wqkv, wqkvT, 4096, 12288);
  // Wo [4096][4096] -> [4096][4096] bf16 transposed
  tconv_kernel<<<dim3(64, 64), 256, 0, stream>>>(wo, woT, 4096, 4096);
  // QKV GEMM: M=4096, N=12288, K=4096, scatter epilogue into q/k/v
  gemm_bf16_kernel<1><<<dim3(96, 32), 256, 0, stream>>>(hsB, wqkvT, 4096, 12288, 4096,
                                                        nullptr, qB, kB, vB);
  // RoPE on q,k: B*NH*S*32 = 2M threads
  rope_kernel<<<8192, 256, 0, stream>>>(pid, qB, kB);
  // Flash attention
  attn_kernel<<<dim3(32, 32), 128, 0, stream>>>(qB, kB, vB, ctxB);
  // Output projection: M=4096, N=4096, K=4096, f32 output
  gemm_bf16_kernel<0><<<dim3(32, 32), 256, 0, stream>>>(ctxB, woT, 4096, 4096, 4096,
                                                        out, nullptr, nullptr, nullptr);
}
